// VariationalGCNEncoder_3934190043984
// MI455X (gfx1250) — matrix operand order repaired and checked by reading, not compiled, not run on hardware
//
#include <hip/hip_runtime.h>

typedef __attribute__((ext_vector_type(16))) __bf16 v16bf;
typedef __attribute__((ext_vector_type(8)))  __bf16 v8bf;
typedef __attribute__((ext_vector_type(8)))  float  v8f;

#define NT 8                 // 8 N-tiles of 16 -> 128 output columns per wave
#define WAVES_PER_BLOCK 8    // 8 wave32 per block -> 128 rows per block
#define C1 128               // 2*H feature width (fixed by reference)
#define KMAX 512             // max K across both GEMMs
#define LDSPAD 8             // bf16 pad per LDS row -> 1040B pitch, conflict-free
#define LDSPITCH (KMAX + LDSPAD)

// ---------------- degree / normalization ----------------

__global__ void k_init_deg(float* deg, int n) {
  int i = blockIdx.x * blockDim.x + threadIdx.x;
  if (i < n) deg[i] = 1.0f;  // self-loop contribution
}

__global__ void k_deg_edges(const int* __restrict__ dst, float* deg, int e) {
  int i = blockIdx.x * blockDim.x + threadIdx.x;
  if (i < e)
    __hip_atomic_fetch_add(&deg[dst[i]], 1.0f, __ATOMIC_RELAXED,
                           __HIP_MEMORY_SCOPE_AGENT);
}

__global__ void k_dinv(float* deg, int n) {
  int i = blockIdx.x * blockDim.x + threadIdx.x;
  if (i < n) {
    float d = deg[i];
    deg[i] = (d > 0.0f) ? rsqrtf(d) : 0.0f;
  }
}

// ---------------- bf16 packing ----------------

__global__ void k_cvt_bf16(const float* __restrict__ s, __bf16* __restrict__ d,
                           size_t n) {
  size_t i = (size_t)blockIdx.x * blockDim.x + threadIdx.x;
  if (i < n) d[i] = (__bf16)s[i];
}

// Wt[j*K + k] = W[k*Ncols + j]   (store B transposed so B columns are contiguous)
__global__ void k_pack_wt(const float* __restrict__ W, __bf16* __restrict__ Wt,
                          int K, int Ncols) {
  int idx = blockIdx.x * blockDim.x + threadIdx.x;
  if (idx < K * Ncols) {
    int j = idx / K, k = idx - j * K;
    Wt[(size_t)j * K + k] = (__bf16)W[(size_t)k * Ncols + j];
  }
}

// Fused [Wmu | Wls] transposed:  Wt[j*K + k], j<H from Wmu, j>=H from Wls
__global__ void k_pack_w2t(const float* __restrict__ Wmu,
                           const float* __restrict__ Wls,
                           __bf16* __restrict__ Wt, int K, int H) {
  int idx = blockIdx.x * blockDim.x + threadIdx.x;
  if (idx < K * 2 * H) {
    int j = idx / K, k = idx - j * K;
    float v = (j < H) ? Wmu[(size_t)k * H + j] : Wls[(size_t)k * H + (j - H)];
    Wt[(size_t)j * K + k] = (__bf16)v;
  }
}

// ---------------- WMMA GEMM: C[M,128] = A[M,K](bf16) x B[K,128](bf16, Bt rows) ----------------
// Bt panel staged once into LDS (padded pitch, bank-conflict free).
// A-fragments ping-pong (a0/a1) across a K-by-64 unrolled loop (no reg-copy rotate).
// B-fragments have one-tile lookahead so LDS latency hides under the WMMA pipe.
// Requires K % 64 == 0 (true for both GEMMs: 512 and 128).

__global__ void k_wmma_gemm(const __bf16* __restrict__ A,
                            const __bf16* __restrict__ Bt,   // [128, K] = B^T
                            float* __restrict__ C, int M, int K) {
  __shared__ __bf16 ldsB[C1 * LDSPITCH];     // 130 KB static LDS (2 blocks/WGP)

  const int tid  = threadIdx.x;
  const int lane = tid & 31;
  const int wave = tid >> 5;

  // cooperative stage: Bt[128][K] -> ldsB[128][LDSPITCH], 16B chunks
  const int chunks = (C1 * K) >> 3;          // v8bf chunks
  for (int cidx = tid; cidx < chunks; cidx += blockDim.x) {
    int elem = cidx << 3;
    int col  = elem / K;
    int kk   = elem - col * K;
    *(v8bf*)(&ldsB[col * LDSPITCH + kk]) = *(const v8bf*)(Bt + (size_t)col * K + kk);
  }
  __syncthreads();

  const int row0 = (blockIdx.x * WAVES_PER_BLOCK + wave) * 16;
  if (row0 >= M) return;                     // wave-uniform: EXEC stays all-ones

  const int mr  = lane & 15;
  const int hi  = lane >> 4;                 // lane half
  const int k0a = hi ? 8 : 0;                // A layout: half-lane K offset
  const int k0b = hi ? 8 : 0;                // B layout: half-lane K offset

  const __bf16* arow = A + (size_t)(row0 + mr) * K;

  auto bfrag = [&](int t, int kk) -> v16bf {
    union { v16bf v; v8bf h[2]; } b;
    const __bf16* p = &ldsB[(t * 16 + mr) * LDSPITCH + kk + k0b];
    b.h[0] = *(const v8bf*)(p);
    b.h[1] = *(const v8bf*)(p + 16);
    return b.v;
  };

  v8f acc[NT] = {};

  union AV { v16bf v; v8bf h[2]; };
  AV a0, a1;
  a0.h[0] = *(const v8bf*)(arow + k0a);
  a0.h[1] = *(const v8bf*)(arow + k0a + 16);
  v16bf bc = bfrag(0, 0);                    // first B fragment pre-loaded

  for (int k = 0; k < K; k += 64) {
    // A fragment for k+32 (always exists: K % 64 == 0)
    a1.h[0] = *(const v8bf*)(arow + k + 32 + k0a);
    a1.h[1] = *(const v8bf*)(arow + k + 32 + k0a + 16);
    __builtin_prefetch(arow + k + 64, 0, 0);
#pragma unroll
    for (int t = 0; t < NT; ++t) {
      v16bf bn = (t < NT - 1) ? bfrag(t + 1, k) : bfrag(0, k + 32);
      acc[t] = __builtin_amdgcn_wmma_f32_16x16x32_bf16(
          false, a0.v, false, bc, (short)0, acc[t], false, false);
      bc = bn;
    }
    // A fragment for k+64
    if (k + 64 < K) {
      a0.h[0] = *(const v8bf*)(arow + k + 64 + k0a);
      a0.h[1] = *(const v8bf*)(arow + k + 64 + k0a + 16);
      __builtin_prefetch(arow + k + 96, 0, 0);
    }
#pragma unroll
    for (int t = 0; t < NT; ++t) {
      v16bf bn = (t < NT - 1) ? bfrag(t + 1, k + 32)
                              : bfrag(0, (k + 64 < K) ? k + 64 : 0);
      acc[t] = __builtin_amdgcn_wmma_f32_16x16x32_bf16(
          false, a1.v, false, bc, (short)0, acc[t], false, false);
      bc = bn;
    }
  }

  const int crow0 = row0 + hi * 8;           // C layout: vgpr r = row crow0+r
#pragma unroll
  for (int t = 0; t < NT; ++t) {
#pragma unroll
    for (int r = 0; r < 8; ++r) {
      int row = crow0 + r;
      if (row < M) C[(size_t)row * C1 + t * 16 + mr] = acc[t][r];
    }
  }
}

// ---------------- graph aggregation ----------------

// out[i,:] = H[i,:] * dinv[i]^2   (self-loop term doubles as accumulator init)
__global__ void k_self_init(const float* __restrict__ Hm,
                            const float* __restrict__ dinv,
                            float* __restrict__ out, int n) {
  size_t idx = (size_t)blockIdx.x * blockDim.x + threadIdx.x;
  if (idx < (size_t)n * C1) {
    int i = (int)(idx >> 7);
    float di = dinv[i];
    out[idx] = Hm[idx] * di * di;
  }
}

// one edge per wave32: 4 features per lane, float4 gather + 4 f32 atomics
__global__ void k_agg_edges(const float* __restrict__ Hm,
                            const int* __restrict__ src,
                            const int* __restrict__ dst,
                            const float* __restrict__ dinv,
                            float* __restrict__ out, int e) {
  int eid = blockIdx.x * (blockDim.x >> 5) + (threadIdx.x >> 5);
  if (eid >= e) return;
  int lane = threadIdx.x & 31;
  int s = src[eid], d = dst[eid];
  float nrm = dinv[s] * dinv[d];
  const float4 v = *(const float4*)(Hm + (size_t)s * C1 + lane * 4);
  float* o = out + (size_t)d * C1 + lane * 4;
  __hip_atomic_fetch_add(o + 0, v.x * nrm, __ATOMIC_RELAXED, __HIP_MEMORY_SCOPE_AGENT);
  __hip_atomic_fetch_add(o + 1, v.y * nrm, __ATOMIC_RELAXED, __HIP_MEMORY_SCOPE_AGENT);
  __hip_atomic_fetch_add(o + 2, v.z * nrm, __ATOMIC_RELAXED, __HIP_MEMORY_SCOPE_AGENT);
  __hip_atomic_fetch_add(o + 3, v.w * nrm, __ATOMIC_RELAXED, __HIP_MEMORY_SCOPE_AGENT);
}

// h = relu(agg + b1) converted to bf16 for the next GEMM
__global__ void k_bias_relu_bf16(const float* __restrict__ agg,
                                 const float* __restrict__ b,
                                 __bf16* __restrict__ outb, int n) {
  size_t idx = (size_t)blockIdx.x * blockDim.x + threadIdx.x;
  if (idx < (size_t)n * C1) {
    int c = (int)(idx & (C1 - 1));
    float v = agg[idx] + b[c];
    outb[idx] = (__bf16)(v > 0.0f ? v : 0.0f);
  }
}

// split fused [mu | logstd] accumulator into d_out = [mu (N*H) ; logstd (N*H)] + biases
__global__ void k_final(const float* __restrict__ agg,
                        const float* __restrict__ bmu,
                        const float* __restrict__ bls,
                        float* __restrict__ out, int n) {
  size_t idx = (size_t)blockIdx.x * blockDim.x + threadIdx.x;
  if (idx < (size_t)n * C1) {
    int i = (int)(idx >> 7);
    int c = (int)(idx & (C1 - 1));
    float v = agg[idx];
    if (c < 64) out[(size_t)i * 64 + c] = v + bmu[c];
    else        out[(size_t)(n + i) * 64 + (c - 64)] = v + bls[c - 64];
  }
}

// ---------------- launcher ----------------

extern "C" void kernel_launch(void* const* d_in, const int* in_sizes, int n_in,
                              void* d_out, int out_size, void* d_ws, size_t ws_size,
                              hipStream_t stream) {
  const float* x   = (const float*)d_in[0];
  const int*   ei  = (const int*)d_in[1];
  const float* W1  = (const float*)d_in[2];
  const float* b1  = (const float*)d_in[3];
  const float* Wmu = (const float*)d_in[4];
  const float* bmu = (const float*)d_in[5];
  const float* Wls = (const float*)d_in[6];
  const float* bls = (const float*)d_in[7];
  float* out = (float*)d_out;

  const int c1 = in_sizes[3];               // 128
  const int F  = in_sizes[2] / c1;          // 512
  const int Nn = in_sizes[0] / F;           // 100000
  const int Ee = in_sizes[1] / 2;           // 1600000
  const int Hh = in_sizes[5];               // 64
  (void)n_in; (void)out_size; (void)ws_size;

  const int* srcp = ei;
  const int* dstp = ei + Ee;

  char* ws = (char*)d_ws;
  size_t off = 0;
  auto alloc = [&](size_t bytes) -> char* {
    char* p = ws + off;
    off += (bytes + 255) & ~(size_t)255;
    return p;
  };
  float*  dinv = (float*) alloc((size_t)Nn * 4);
  __bf16* xb   = (__bf16*)alloc((size_t)Nn * F * 2);
  __bf16* w1t  = (__bf16*)alloc((size_t)F * c1 * 2);
  float*  p1   = (float*) alloc((size_t)Nn * c1 * 4);
  float*  agg1 = (float*) alloc((size_t)Nn * c1 * 4);
  __bf16* hb   = (__bf16*)alloc((size_t)Nn * c1 * 2);
  __bf16* w2t  = (__bf16*)alloc((size_t)c1 * c1 * 2);
  float*  p2   = (float*) alloc((size_t)Nn * c1 * 4);
  float*  agg2 = (float*) alloc((size_t)Nn * c1 * 4);

  const int TPB = 256;
  auto blks  = [&](size_t work) { return (unsigned)((work + TPB - 1) / TPB); };

  // graph normalization
  k_init_deg <<<blks(Nn), TPB, 0, stream>>>(dinv, Nn);
  k_deg_edges<<<blks(Ee), TPB, 0, stream>>>(dstp, dinv, Ee);
  k_dinv     <<<blks(Nn), TPB, 0, stream>>>(dinv, Nn);

  // bf16 packing
  k_cvt_bf16 <<<blks((size_t)Nn * F), TPB, 0, stream>>>(x, xb, (size_t)Nn * F);
  k_pack_wt  <<<blks((size_t)F * c1), TPB, 0, stream>>>(W1, w1t, F, c1);
  k_pack_w2t <<<blks((size_t)c1 * c1), TPB, 0, stream>>>(Wmu, Wls, w2t, c1, Hh);

  const unsigned gblocks = (unsigned)((Nn + 16 * WAVES_PER_BLOCK - 1) / (16 * WAVES_PER_BLOCK));
  const unsigned eblocks = (unsigned)((Ee + (TPB / 32) - 1) / (TPB / 32));

  // conv1: transform -> aggregate -> bias+relu (+bf16)
  k_wmma_gemm     <<<gblocks, 32 * WAVES_PER_BLOCK, 0, stream>>>(xb, w1t, p1, Nn, F);
  k_self_init     <<<blks((size_t)Nn * c1), TPB, 0, stream>>>(p1, dinv, agg1, Nn);
  k_agg_edges     <<<eblocks, TPB, 0, stream>>>(p1, srcp, dstp, dinv, agg1, Ee);
  k_bias_relu_bf16<<<blks((size_t)Nn * c1), TPB, 0, stream>>>(agg1, b1, hb, Nn);

  // conv2+conv3 fused: transform -> aggregate -> split mu/logstd
  k_wmma_gemm     <<<gblocks, 32 * WAVES_PER_BLOCK, 0, stream>>>(hb, w2t, p2, Nn, c1);
  k_self_init     <<<blks((size_t)Nn * c1), TPB, 0, stream>>>(p2, dinv, agg2, Nn);
  k_agg_edges     <<<eblocks, TPB, 0, stream>>>(p2, srcp, dstp, dinv, agg2, Ee);
  k_final         <<<blks((size_t)Nn * c1), TPB, 0, stream>>>(agg2, bmu, bls, out, Nn);
}
